// TF_RNN_GRU_CHS_24309514895948
// MI455X (gfx1250) — matrix operand order repaired and checked by reading, not compiled, not run on hardware
//
#include <hip/hip_runtime.h>

// ---------------------------------------------------------------------------
// 3-layer GRU, batch=32, hidden=512, 999 steps, bf16 WMMA w/ fp32 accumulate.
// gfx1250 (CDNA5): v_wmma_f32_16x16x32_bf16, wave32.
// ---------------------------------------------------------------------------

typedef __attribute__((ext_vector_type(16))) __bf16 v16bf;
typedef __attribute__((ext_vector_type(8)))  __bf16 v8bf;
typedef __attribute__((ext_vector_type(8)))  float  v8f;

#define BATCH   32
#define HID     512
#define GATES   1536
#define DDIM    128
#define TSTEPS  999
#define OUTS    100

__device__ __forceinline__ v8f wmma_bf16(v16bf a, v16bf b, v8f c) {
  // D = A(16x32 bf16) * B(32x16 bf16) + C(16x16 f32)
  return __builtin_amdgcn_wmma_f32_16x16x32_bf16(
      /*neg_a=*/false, a, /*neg_b=*/false, b,
      /*c_mod=*/(short)0, c, /*reuse_a=*/false, /*reuse_b=*/false);
}

// A-fragment (16x32, M x K) from a row-major bf16 matrix with leading dim ld.
// ISA layout: lane m=lane&15; lanes0-15 hold K {0..7,16..23}, lanes16-31 {8..15,24..31}.
__device__ __forceinline__ v16bf load_a_frag(const __bf16* __restrict__ A, int ld,
                                             int m0, int kt, int lane) {
  const int m  = m0 + (lane & 15);
  const int k0 = (kt << 5) + ((lane >> 4) << 3);
  const __bf16* p = A + (size_t)m * ld + k0;
  v8bf lo = *(const v8bf*)p;           // K = k0..k0+7
  v8bf hi = *(const v8bf*)(p + 16);    // K = k0+16..k0+23
  v16bf r;
#pragma unroll
  for (int i = 0; i < 8; ++i) { r[i] = lo[i]; r[i + 8] = hi[i]; }
  return r;
}

// B-fragment (32x16, K x N) from pre-packed buffer: tile (kt,nt) stores the 16
// bf16 each lane needs contiguously (lane l -> B[kt*32+(l>>4)*8 + {0..7,16..23}][nt*16+(l&15)]).
__device__ __forceinline__ v16bf load_b_frag(const __bf16* __restrict__ Bp, int ktc,
                                             int nt, int kt, int lane) {
  const __bf16* p = Bp + (((size_t)(nt * ktc + kt)) * 32 + lane) * 16;
  v8bf lo = *(const v8bf*)p;
  v8bf hi = *(const v8bf*)(p + 8);
  v16bf r;
#pragma unroll
  for (int i = 0; i < 8; ++i) { r[i] = lo[i]; r[i + 8] = hi[i]; }
  return r;
}

__device__ __forceinline__ float sigf(float x) { return 1.0f / (1.0f + __expf(-x)); }

// ---------------------------------------------------------------------------
// One GRU layer for one timestep.  grid=16 blocks x 128 thr (4 waves); one wave
// per 16x16 tile of the 32x512 hidden state (64 tiles).
// ---------------------------------------------------------------------------
template <int KTC_IN>
__global__ __launch_bounds__(128) void gru_layer_kernel(
    const __bf16* __restrict__ Ain, int ldA,        // layer input (32 x 32*KTC_IN) bf16
    const __bf16* __restrict__ Wp,                  // packed (32*KTC_IN) x 1536
    const __bf16* __restrict__ Up,                  // packed 512 x 1536
    const float*  __restrict__ bias,                // (2,1536) fp32
    const __bf16* __restrict__ hbp,                 // prev h, bf16 32x512
    float*        __restrict__ h,                   // fp32 state 32x512 (in/out)
    __bf16*       __restrict__ hbn,                 // new h bf16 32x512
    float*        __restrict__ hid,                 // d_out hidden slice (+j*512) or null
    __bf16*       __restrict__ yb,                  // bf16 rows for prediction or null
    int sidx)                                       // output step index (t-899)
{
  const int lane = threadIdx.x & 31;
  const int tile = blockIdx.x * 4 + (threadIdx.x >> 5);  // 0..63
  const int mt   = tile >> 5;                            // 0..1  (batch tile)
  const int nt   = tile & 31;                            // 0..31 (hidden tile)

  v8f ax0 = {}, ax1 = {}, ax2 = {};   // x-path: z, r, h gates
  v8f ah0 = {}, ah1 = {}, ah2 = {};   // h-path: z, r, h gates

  // mx = Ain @ W   (three gate column-blocks: nt, nt+32, nt+64 of 96 n-tiles)
#pragma unroll
  for (int kt = 0; kt < KTC_IN; ++kt) {
    v16bf a = load_a_frag(Ain, ldA, mt * 16, kt, lane);
    ax0 = wmma_bf16(a, load_b_frag(Wp, KTC_IN, nt,      kt, lane), ax0);
    ax1 = wmma_bf16(a, load_b_frag(Wp, KTC_IN, nt + 32, kt, lane), ax1);
    ax2 = wmma_bf16(a, load_b_frag(Wp, KTC_IN, nt + 64, kt, lane), ax2);
  }
  // mh = h_prev @ U  (K = 512 -> 16 k-tiles)
#pragma unroll 4
  for (int kt = 0; kt < 16; ++kt) {
    v16bf a = load_a_frag(hbp, HID, mt * 16, kt, lane);
    ah0 = wmma_bf16(a, load_b_frag(Up, 16, nt,      kt, lane), ah0);
    ah1 = wmma_bf16(a, load_b_frag(Up, 16, nt + 32, kt, lane), ah1);
    ah2 = wmma_bf16(a, load_b_frag(Up, 16, nt + 64, kt, lane), ah2);
  }

  const int   n   = nt * 16 + (lane & 15);
  const float bxz = bias[n],            bxr = bias[512 + n],        bxh = bias[1024 + n];
  const float bhz = bias[1536 + n],     bhr = bias[1536 + 512 + n], bhh = bias[1536 + 1024 + n];

#pragma unroll
  for (int r = 0; r < 8; ++r) {
    const int   m    = mt * 16 + ((lane >> 4) << 3) + r;   // C layout: VGPR r -> M row
    const float z    = sigf(ax0[r] + bxz + ah0[r] + bhz);
    const float rg   = sigf(ax1[r] + bxr + ah1[r] + bhr);
    const float cand = tanhf(ax2[r] + bxh + rg * (ah2[r] + bhh));
    const float hold = h[m * HID + n];
    const float hn   = z * hold + (1.0f - z) * cand;
    h[m * HID + n]   = hn;
    hbn[m * HID + n] = (__bf16)hn;
    if (hid) hid[((size_t)(m * OUTS + sidx) * 3) * HID + n] = hn;
    if (yb)  yb[((size_t)(sidx * BATCH + m)) * HID + n]     = (__bf16)hn;
  }
}

// predictions = ys(3200x512) @ Wd(512x128) + bd ; 200 M-tiles x 8 N-tiles.
__global__ __launch_bounds__(128) void pred_kernel(
    const __bf16* __restrict__ yb, const __bf16* __restrict__ Wpd,
    const float* __restrict__ bd, float* __restrict__ out)
{
  const int lane = threadIdx.x & 31;
  const int tile = blockIdx.x * 4 + (threadIdx.x >> 5);  // 0..1599
  const int mt   = tile >> 3;
  const int nt   = tile & 7;
  v8f acc = {};
#pragma unroll 4
  for (int kt = 0; kt < 16; ++kt) {
    v16bf a = load_a_frag(yb, HID, mt * 16, kt, lane);
    acc = wmma_bf16(a, load_b_frag(Wpd, 16, nt, kt, lane), acc);
  }
  const int   n  = nt * 16 + (lane & 15);
  const float bv = bd[n];
#pragma unroll
  for (int r = 0; r < 8; ++r) {
    const int rr = mt * 16 + ((lane >> 4) << 3) + r;  // row = s*32+m
    const int s  = rr >> 5, m = rr & 31;
    out[((size_t)(m * OUTS + s)) * DDIM + n] = acc[r] + bv;
  }
}

// ------------------------ setup kernels ------------------------------------
__global__ void cvt_bf16_kernel(const float* __restrict__ src, __bf16* __restrict__ dst, int n) {
  int i = blockIdx.x * blockDim.x + threadIdx.x;
  if (i < n) dst[i] = (__bf16)src[i];
}

// Pack row-major KxN fp32 weight into WMMA B-fragment order (bf16).
__global__ void pack_b_kernel(const float* __restrict__ src, __bf16* __restrict__ dst,
                              int K, int N) {
  int idx = blockIdx.x * blockDim.x + threadIdx.x;
  if (idx >= K * N) return;
  int tile = idx >> 9, within = idx & 511;
  int lane = within >> 4, i = within & 15;
  int ktc = K >> 5;
  int nt = tile / ktc, kt = tile - nt * ktc;
  int k = kt * 32 + ((lane >> 4) << 3) + (i & 7) + ((i >> 3) << 4);
  int n = nt * 16 + (lane & 15);
  dst[idx] = (__bf16)src[k * N + n];
}

__global__ void init_h_kernel(const float* __restrict__ s, float* __restrict__ h,
                              __bf16* __restrict__ hb) {
  int i = blockIdx.x * blockDim.x + threadIdx.x;
  if (i >= BATCH * HID) return;
  float v = s[i & (HID - 1)];
  h[i] = v;
  hb[i] = (__bf16)v;
}

// ---------------------------------------------------------------------------
extern "C" void kernel_launch(void* const* d_in, const int* in_sizes, int n_in,
                              void* d_out, int out_size, void* d_ws, size_t ws_size,
                              hipStream_t stream) {
  (void)in_sizes; (void)n_in; (void)out_size; (void)ws_size;
  const float* x     = (const float*)d_in[0];
  const float* W[3]  = {(const float*)d_in[1], (const float*)d_in[5], (const float*)d_in[9]};
  const float* U[3]  = {(const float*)d_in[2], (const float*)d_in[6], (const float*)d_in[10]};
  const float* bb[3] = {(const float*)d_in[3], (const float*)d_in[7], (const float*)d_in[11]};
  const float* sv[3] = {(const float*)d_in[4], (const float*)d_in[8], (const float*)d_in[12]};
  const float* Wd    = (const float*)d_in[13];
  const float* bd    = (const float*)d_in[14];

  char*  ws  = (char*)d_ws;
  size_t off = 0;
  auto alloc = [&](size_t bytes) -> char* {
    char* p = ws + off;
    off = (off + bytes + 255) & ~(size_t)255;
    return p;
  };

  __bf16* xb = (__bf16*)alloc((size_t)BATCH * 1000 * DDIM * 2);
  __bf16* Wp[3]; __bf16* Up[3];
  Wp[0] = (__bf16*)alloc((size_t)DDIM * GATES * 2);
  Up[0] = (__bf16*)alloc((size_t)HID * GATES * 2);
  Wp[1] = (__bf16*)alloc((size_t)HID * GATES * 2);
  Up[1] = (__bf16*)alloc((size_t)HID * GATES * 2);
  Wp[2] = (__bf16*)alloc((size_t)HID * GATES * 2);
  Up[2] = (__bf16*)alloc((size_t)HID * GATES * 2);
  __bf16* Wpd = (__bf16*)alloc((size_t)HID * DDIM * 2);
  float*  hst[3]; __bf16* hb[3][2];
  for (int j = 0; j < 3; ++j) {
    hst[j]   = (float*)alloc((size_t)BATCH * HID * 4);
    hb[j][0] = (__bf16*)alloc((size_t)BATCH * HID * 2);
    hb[j][1] = (__bf16*)alloc((size_t)BATCH * HID * 2);
  }
  __bf16* yb = (__bf16*)alloc((size_t)OUTS * BATCH * HID * 2);

  // x -> bf16 (layout unchanged: (32,1000,128); per-step base t*128, ld=128000)
  {
    int n = BATCH * 1000 * DDIM;
    cvt_bf16_kernel<<<(n + 255) / 256, 256, 0, stream>>>(x, xb, n);
  }
  // Pack weights to B-fragment order
  pack_b_kernel<<<(DDIM * GATES + 255) / 256, 256, 0, stream>>>(W[0], Wp[0], DDIM, GATES);
  pack_b_kernel<<<(HID * GATES + 255) / 256, 256, 0, stream>>>(U[0], Up[0], HID, GATES);
  pack_b_kernel<<<(HID * GATES + 255) / 256, 256, 0, stream>>>(W[1], Wp[1], HID, GATES);
  pack_b_kernel<<<(HID * GATES + 255) / 256, 256, 0, stream>>>(U[1], Up[1], HID, GATES);
  pack_b_kernel<<<(HID * GATES + 255) / 256, 256, 0, stream>>>(W[2], Wp[2], HID, GATES);
  pack_b_kernel<<<(HID * GATES + 255) / 256, 256, 0, stream>>>(U[2], Up[2], HID, GATES);
  pack_b_kernel<<<(HID * DDIM + 255) / 256, 256, 0, stream>>>(Wd, Wpd, HID, DDIM);
  for (int j = 0; j < 3; ++j)
    init_h_kernel<<<(BATCH * HID + 255) / 256, 256, 0, stream>>>(sv[j], hst[j], hb[j][0]);

  float* hid_base = (float*)d_out + (size_t)BATCH * OUTS * DDIM;

  for (int t = 0; t < TSTEPS; ++t) {
    const int  p    = t & 1;
    const int  sidx = t - (TSTEPS - OUTS);  // t - 899
    const bool emit = (sidx >= 0);
    // layer 0 (input x_t, K=128)
    gru_layer_kernel<4><<<16, 128, 0, stream>>>(
        xb + (size_t)t * DDIM, 1000 * DDIM, Wp[0], Up[0], bb[0],
        hb[0][p], hst[0], hb[0][p ^ 1],
        emit ? hid_base + 0 * HID : nullptr, nullptr, sidx);
    // layer 1 (input = new h0, K=512)
    gru_layer_kernel<16><<<16, 128, 0, stream>>>(
        hb[0][p ^ 1], HID, Wp[1], Up[1], bb[1],
        hb[1][p], hst[1], hb[1][p ^ 1],
        emit ? hid_base + 1 * HID : nullptr, nullptr, sidx);
    // layer 2 (input = new h1, K=512); also emit bf16 rows for prediction GEMM
    gru_layer_kernel<16><<<16, 128, 0, stream>>>(
        hb[1][p ^ 1], HID, Wp[2], Up[2], bb[2],
        hb[2][p], hst[2], hb[2][p ^ 1],
        emit ? hid_base + 2 * HID : nullptr, emit ? yb : nullptr, sidx);
  }

  // predictions = ys @ Wd + bd
  pred_kernel<<<400, 128, 0, stream>>>(yb, Wpd, bd, (float*)d_out);
}
